// GraphPolicy_31233002177041
// MI455X (gfx1250) — matrix operand order repaired and checked by reading, not compiled, not run on hardware
//
#include <hip/hip_runtime.h>
#include <hip/hip_bf16.h>

typedef __bf16 bf16;
typedef __attribute__((ext_vector_type(16))) __bf16 v16bf;
typedef __attribute__((ext_vector_type(8)))  __bf16 v8bf;
typedef __attribute__((ext_vector_type(2)))  __bf16 v2bf;
typedef __attribute__((ext_vector_type(8)))  float  v8f;

#define NN 8192
#define KSPLIT 8
#define NKB (NN / 32)          // 256 k-blocks of 32
#define LKB (NKB / KSPLIT)     // 32 k-blocks per grid.z slice

__device__ __forceinline__ float leaky(float v) { return v >= 0.0f ? v : 0.01f * v; }
__device__ __forceinline__ float sigmoidf(float v) { return 1.0f / (1.0f + expf(-v)); }

// ---------------------------------------------------------------------------
// Kernel 1: one NT sweep of f32 adj -> per-tile row/col partial sums (determ.),
//           bf16 adj16 (row-major) and bf16 adjT16 (transposed, row-major).
// 128x128 tile staged in LDS (pad 129 to dodge bank conflicts).
// ---------------------------------------------------------------------------
__global__ void prep_kernel(const float* __restrict__ adj,
                            bf16* __restrict__ a16, bf16* __restrict__ at16,
                            float* __restrict__ rowpart, float* __restrict__ colpart) {
    extern __shared__ float tile[];   // 128 * 129 floats
    const int tr0 = blockIdx.y * 128, tc0 = blockIdx.x * 128;
    const int t = threadIdx.x;        // 256 threads

    #pragma unroll 4
    for (int i = 0; i < 64; ++i) {
        int r = i * 2 + (t >> 7);
        int cc = t & 127;
        tile[r * 129 + cc] =
            __builtin_nontemporal_load(&adj[(size_t)(tr0 + r) * NN + tc0 + cc]);
    }
    __syncthreads();

    if (t < 128) {                    // waves 0..3: row partial sums (per col-tile)
        float s = 0.0f;
        for (int cc = 0; cc < 128; ++cc) s += fabsf(tile[t * 129 + cc]);
        rowpart[(size_t)blockIdx.x * NN + tr0 + t] = s;
    } else {                          // waves 4..7: col partial sums (per row-tile)
        int cc = t - 128;
        float s = 0.0f;
        for (int r = 0; r < 128; ++r) s += fabsf(tile[r * 129 + cc]);
        colpart[(size_t)blockIdx.y * NN + tc0 + cc] = s;
    }

    // adj16 row-major, 2 bf16 per store
    #pragma unroll 4
    for (int i = 0; i < 32; ++i) {
        int r = i * 4 + (t >> 6);
        int c2 = (t & 63) * 2;
        v2bf p; p[0] = (bf16)tile[r * 129 + c2]; p[1] = (bf16)tile[r * 129 + c2 + 1];
        *(v2bf*)(a16 + (size_t)(tr0 + r) * NN + tc0 + c2) = p;
    }
    // adjT16: out row = adj column, out col = adj row
    #pragma unroll 4
    for (int i = 0; i < 32; ++i) {
        int cr = i * 4 + (t >> 6);    // adj col within tile
        int rr = (t & 63) * 2;        // adj row pair
        v2bf p; p[0] = (bf16)tile[rr * 129 + cr]; p[1] = (bf16)tile[(rr + 1) * 129 + cr];
        *(v2bf*)(at16 + (size_t)(tc0 + cr) * NN + tr0 + rr) = p;
    }
}

// Deterministic fixed-order reduction of 64 tile partials per index.
__global__ void sum64_kernel(const float* __restrict__ part, float* __restrict__ out) {
    int i = blockIdx.x * blockDim.x + threadIdx.x;
    float s = 0.0f;
    #pragma unroll 8
    for (int b = 0; b < 64; ++b) s += part[(size_t)b * NN + i];
    out[i] = s;
}

// ---------------------------------------------------------------------------
// Pack RHS [NN, F] f32 into the WMMA B register layout (bf16 panel):
// panel[((kb*nh + h)*32 + lane)*16 + s] = x[kb*32 + (lane>=16?8:0)+(s&7)+(s>=8?16:0)][h*16+(lane&15)]
// ---------------------------------------------------------------------------
__global__ void pack_panel(const float* __restrict__ src, bf16* __restrict__ dst, int F) {
    const int nh = F >> 4;
    size_t idx = (size_t)blockIdx.x * blockDim.x + threadIdx.x;
    if (idx >= (size_t)NN * F) return;
    int s    = idx & 15;
    int lane = (idx >> 4) & 31;
    int tt   = (int)(idx >> 9);
    int h    = tt % nh;
    int kb   = tt / nh;
    int k    = kb * 32 + ((lane >> 4) << 3) + (s & 7) + ((s >> 3) << 4);
    int f    = h * 16 + (lane & 15);
    dst[idx] = (bf16)src[(size_t)k * F + f];
}

// ---------------------------------------------------------------------------
// Core WMMA GEMM, K-split across grid.z with raw partial outputs.
// B panel slice staged in LDS; A streamed NT with depth-2 register pipeline.
// 256 threads = 8 waves; 16 rows/wave; grid.y selects A vs A^T side.
// ---------------------------------------------------------------------------
template <int NHALF>
__global__ void gemm_kernel(const bf16* __restrict__ matA, const bf16* __restrict__ matB,
                            const bf16* __restrict__ panel,
                            float* __restrict__ poutA, float* __restrict__ poutB) {
    extern __shared__ bf16 bslice[];                 // LKB * NHALF KB
    const int F = NHALF * 16;
    const int z = blockIdx.z;
    const int kb0 = z * LKB;
    const bf16* mat = (blockIdx.y == 0) ? matA : matB;
    float* pout = ((blockIdx.y == 0) ? poutA : poutB) + (size_t)z * NN * F;

    // cooperative copy of this K-slice of the B panel into LDS
    {
        const size_t sliceElems = (size_t)LKB * NHALF * 32 * 16;
        const bf16* psrc = panel + (size_t)kb0 * NHALF * 32 * 16;
        for (size_t e = (size_t)threadIdx.x * 8; e < sliceElems; e += (size_t)blockDim.x * 8)
            *(v8bf*)(bslice + e) = *(const v8bf*)(psrc + e);
    }
    __syncthreads();

    const int lane = threadIdx.x & 31;
    const int wave = threadIdx.x >> 5;
    const int r0   = (blockIdx.x * 8 + wave) * 16;
    // 16-bit A-matrix layout: lanes 0-15 -> M=lane, K {0..7,16..23};
    //                         lanes 16-31 -> M=lane-16, K {8..15,24..31}
    const int row    = r0 + (lane & 15);
    const int kshift = (lane >> 4) ? 8 : 0;
    const bf16* arow = mat + (size_t)row * NN + kb0 * 32 + kshift;

    v8f c[NHALF];
    #pragma unroll
    for (int h = 0; h < NHALF; ++h) c[h] = (v8f){0.f,0.f,0.f,0.f,0.f,0.f,0.f,0.f};

    // depth-2 A prefetch pipeline (branchless clamped tail)
    v8bf alo0 = __builtin_nontemporal_load((const v8bf*)(arow));
    v8bf ahi0 = __builtin_nontemporal_load((const v8bf*)(arow + 16));
    v8bf alo1 = __builtin_nontemporal_load((const v8bf*)(arow + 32));
    v8bf ahi1 = __builtin_nontemporal_load((const v8bf*)(arow + 48));

    for (int lkb = 0; lkb < LKB; ++lkb) {
        v16bf a = __builtin_shufflevector(alo0, ahi0, 0,1,2,3,4,5,6,7,8,9,10,11,12,13,14,15);
        alo0 = alo1; ahi0 = ahi1;
        int kn = (lkb + 2 < LKB) ? (lkb + 2) : (LKB - 1);
        alo1 = __builtin_nontemporal_load((const v8bf*)(arow + kn * 32));
        ahi1 = __builtin_nontemporal_load((const v8bf*)(arow + kn * 32 + 16));
        #pragma unroll
        for (int h = 0; h < NHALF; ++h) {
            v16bf b = *(const v16bf*)(bslice + (((size_t)lkb * NHALF + h) * 32 + lane) * 16);
            c[h] = __builtin_amdgcn_wmma_f32_16x16x32_bf16(
                       false, a, false, b, (short)0, c[h], false, false);
        }
    }

    #pragma unroll
    for (int r = 0; r < 8; ++r) {
        int m = r0 + r + ((lane >> 4) ? 8 : 0);     // C/D layout: vgpr r, lanes>=16 -> M+8
        #pragma unroll
        for (int h = 0; h < NHALF; ++h)
            pout[(size_t)m * F + h * 16 + (lane & 15)] = c[h][r];
    }
}

// Deterministic combine of KSPLIT partials + L1-norm scale.
__global__ void reduce_scale(const float* __restrict__ pA, const float* __restrict__ pB,
                             const float* __restrict__ sumA, const float* __restrict__ sumB,
                             float* __restrict__ outA, float* __restrict__ outB, int F) {
    const float* p   = (blockIdx.y == 0) ? pA : pB;
    const float* sum = (blockIdx.y == 0) ? sumA : sumB;
    float*       out = (blockIdx.y == 0) ? outA : outB;
    size_t idx = (size_t)blockIdx.x * blockDim.x + threadIdx.x;   // over NN*F
    size_t stride = (size_t)NN * F;
    float s = 0.0f;
    #pragma unroll
    for (int z = 0; z < KSPLIT; ++z) s += p[idx + (size_t)z * stride];
    int m = (int)(idx / F);
    out[idx] = s / fmaxf(sum[m], 1e-12f);
}

// ---------------------------------------------------------------------------
// Layer 1: v1 = leaky([Ax, ATx, x] @ W1^T + b1); per-wave column sums of l1.
// ---------------------------------------------------------------------------
__global__ void layer1_kernel(const float* __restrict__ Ax, const float* __restrict__ ATx,
                              const float* __restrict__ x,
                              const float* __restrict__ W1, const float* __restrict__ b1,
                              float* __restrict__ v1f, float* __restrict__ l1part) {
    __shared__ float w[32 * 96];
    __shared__ float bs[32];
    for (int i = threadIdx.x; i < 32 * 96; i += blockDim.x) w[i] = W1[i];
    if (threadIdx.x < 32) bs[threadIdx.x] = b1[threadIdx.x];
    __syncthreads();

    const int i = blockIdx.x * blockDim.x + threadIdx.x;
    float l1[96];
    #pragma unroll
    for (int j = 0; j < 32; ++j) {
        l1[j]      = Ax [(size_t)i * 32 + j];
        l1[32 + j] = ATx[(size_t)i * 32 + j];
        l1[64 + j] = x  [(size_t)i * 32 + j];
    }
    for (int o = 0; o < 32; ++o) {
        float acc = bs[o];
        #pragma unroll
        for (int j = 0; j < 96; ++j) acc += w[o * 96 + j] * l1[j];
        v1f[(size_t)i * 32 + o] = leaky(acc);
    }
    for (int j = 0; j < 96; ++j) {      // deterministic wave-level column sums
        float v = l1[j];
        #pragma unroll
        for (int off = 16; off > 0; off >>= 1) v += __shfl_xor(v, off, 32);
        if ((threadIdx.x & 31) == 0) l1part[(size_t)(i >> 5) * 96 + j] = v;
    }
}

__global__ void u1_kernel(const float* __restrict__ l1part, const float* __restrict__ uW1,
                          const float* __restrict__ ub1, float* __restrict__ u1) {
    __shared__ float l1s[96];
    int t = threadIdx.x;     // 96 threads
    if (t < 96) {
        float s = 0.0f;
        for (int w = 0; w < NN / 32; ++w) s += l1part[(size_t)w * 96 + t];
        l1s[t] = s;
    }
    __syncthreads();
    if (t < 16) {
        float acc = ub1[t];
        for (int j = 0; j < 96; ++j) acc += uW1[t * 96 + j] * (l1s[j] * (1.0f / NN));
        u1[t] = leaky(acc);
    }
}

// ---------------------------------------------------------------------------
// LSTM: gates = [u1, Av1, ATv1, v1] @ Wih^T + bih + h @ Whh^T + bhh
// LSTM_HID = 32 (i,f,g,o each 32). Weights in dynamic LDS.
// ---------------------------------------------------------------------------
__global__ void lstm_kernel(const float* __restrict__ u1, const float* __restrict__ Av1,
                            const float* __restrict__ ATv1, const float* __restrict__ v1f,
                            const float* __restrict__ hin, const float* __restrict__ cin,
                            const float* __restrict__ Wih, const float* __restrict__ Whh,
                            const float* __restrict__ bih, const float* __restrict__ bhh,
                            float* __restrict__ h_out, float* __restrict__ c_out,
                            float* __restrict__ v2f, float* __restrict__ u2part) {
    extern __shared__ float smem[];
    float* wih  = smem;                 // 128*112
    float* whh  = wih + 128 * 112;      // 128*32
    float* bsum = whh + 128 * 32;       // 128
    float* u1s  = bsum + 128;           // 16
    for (int i = threadIdx.x; i < 128 * 112; i += blockDim.x) wih[i] = Wih[i];
    for (int i = threadIdx.x; i < 128 * 32;  i += blockDim.x) whh[i] = Whh[i];
    for (int i = threadIdx.x; i < 128;       i += blockDim.x) bsum[i] = bih[i] + bhh[i];
    if (threadIdx.x < 16) u1s[threadIdx.x] = u1[threadIdx.x];
    __syncthreads();

    const int i = blockIdx.x * blockDim.x + threadIdx.x;
    float l2[112], hv[32];
    #pragma unroll
    for (int j = 0; j < 16; ++j) l2[j] = u1s[j];
    #pragma unroll
    for (int j = 0; j < 32; ++j) {
        l2[16 + j] = Av1 [(size_t)i * 32 + j];
        l2[48 + j] = ATv1[(size_t)i * 32 + j];
        l2[80 + j] = v1f [(size_t)i * 32 + j];
        hv[j]      = hin [(size_t)i * 32 + j];
    }

    for (int j = 0; j < 32; ++j) {      // gates i,f,g,o = rows j,32+j,64+j,96+j
        float gi = bsum[j], gf = bsum[32 + j], gg = bsum[64 + j], go = bsum[96 + j];
        #pragma unroll
        for (int k = 0; k < 112; ++k) {
            float v = l2[k];
            gi += wih[j * 112 + k] * v;  gf += wih[(32 + j) * 112 + k] * v;
            gg += wih[(64 + j) * 112 + k] * v;  go += wih[(96 + j) * 112 + k] * v;
        }
        #pragma unroll
        for (int k = 0; k < 32; ++k) {
            float v = hv[k];
            gi += whh[j * 32 + k] * v;  gf += whh[(32 + j) * 32 + k] * v;
            gg += whh[(64 + j) * 32 + k] * v;  go += whh[(96 + j) * 32 + k] * v;
        }
        float co = sigmoidf(gf) * cin[(size_t)i * 32 + j] + sigmoidf(gi) * tanhf(gg);
        float ho = sigmoidf(go) * tanhf(co);
        c_out[(size_t)i * 32 + j] = co;
        h_out[(size_t)i * 32 + j] = ho;
        if (j < 16) {
            v2f[(size_t)i * 16 + j] = leaky(ho);
        } else {
            float v = ho;
            #pragma unroll
            for (int off = 16; off > 0; off >>= 1) v += __shfl_xor(v, off, 32);
            if ((threadIdx.x & 31) == 0) u2part[(size_t)(i >> 5) * 16 + (j - 16)] = v;
        }
    }
}

__global__ void u2_kernel(const float* __restrict__ u2part, float* __restrict__ u2f) {
    int g = threadIdx.x;
    if (g < 16) {
        float s = 0.0f;
        for (int w = 0; w < NN / 32; ++w) s += u2part[(size_t)w * 16 + g];
        u2f[g] = leaky(s * (1.0f / NN));
    }
}

// ---------------------------------------------------------------------------
// Threefry2x32 (JAX), key(42) = (0, 42); Gumbel-argmax categorical sampling.
// ---------------------------------------------------------------------------
__device__ __forceinline__ unsigned rotl32(unsigned x, int n) { return (x << n) | (x >> (32 - n)); }
__device__ uint2 threefry2x32(unsigned k0, unsigned k1, unsigned c0, unsigned c1) {
    const unsigned ks2 = k0 ^ k1 ^ 0x1BD11BDAu;
    unsigned x0 = c0 + k0, x1 = c1 + k1;
    const int R0[4] = {13, 15, 26, 6}, R1[4] = {17, 29, 16, 24};
    #pragma unroll
    for (int i = 0; i < 4; ++i) { x0 += x1; x1 = rotl32(x1, R0[i]); x1 ^= x0; }
    x0 += k1;  x1 += ks2 + 1u;
    #pragma unroll
    for (int i = 0; i < 4; ++i) { x0 += x1; x1 = rotl32(x1, R1[i]); x1 ^= x0; }
    x0 += ks2; x1 += k0 + 2u;
    #pragma unroll
    for (int i = 0; i < 4; ++i) { x0 += x1; x1 = rotl32(x1, R0[i]); x1 ^= x0; }
    x0 += k0;  x1 += k1 + 3u;
    #pragma unroll
    for (int i = 0; i < 4; ++i) { x0 += x1; x1 = rotl32(x1, R1[i]); x1 ^= x0; }
    x0 += k1;  x1 += ks2 + 4u;
    #pragma unroll
    for (int i = 0; i < 4; ++i) { x0 += x1; x1 = rotl32(x1, R0[i]); x1 ^= x0; }
    x0 += ks2; x1 += k0 + 5u;
    return make_uint2(x0, x1);
}

__global__ void layer3_kernel(const float* __restrict__ u2f, const float* __restrict__ Av2,
                              const float* __restrict__ ATv2, const float* __restrict__ v2f,
                              const float* __restrict__ W3, const float* __restrict__ b3,
                              float* __restrict__ actions) {
    __shared__ float w3[3 * 64];
    __shared__ float b3s[3];
    __shared__ float u2s[16];
    if (threadIdx.x < 3 * 64) w3[threadIdx.x] = W3[threadIdx.x];
    if (threadIdx.x < 3)      b3s[threadIdx.x] = b3[threadIdx.x];
    if (threadIdx.x < 16)     u2s[threadIdx.x] = u2f[threadIdx.x];
    __syncthreads();

    const int i = blockIdx.x * blockDim.x + threadIdx.x;
    float l3[64];
    #pragma unroll
    for (int g = 0; g < 16; ++g) l3[g] = u2s[g];
    #pragma unroll
    for (int j = 0; j < 16; ++j) {
        l3[16 + j] = Av2 [(size_t)i * 16 + j];
        l3[32 + j] = ATv2[(size_t)i * 16 + j];
        l3[48 + j] = v2f [(size_t)i * 16 + j];
    }

    const unsigned half = (unsigned)(NN * 3 / 2);   // 12288
    float best = -1e30f; int arg = 0;
    #pragma unroll
    for (int k = 0; k < 3; ++k) {
        float acc = b3s[k];
        #pragma unroll
        for (int j = 0; j < 64; ++j) acc += w3[k * 64 + j] * l3[j];
        float logit = 2.0f * tanhf(acc);

        unsigned idx = (unsigned)i * 3u + (unsigned)k;
        unsigned j   = (idx < half) ? idx : idx - half;
        uint2 o = threefry2x32(0u, 42u, j, j + half);
        unsigned bits = (idx < half) ? o.x : o.y;
        const float tiny = 1.17549435e-38f;
        float f = __uint_as_float((bits >> 9) | 0x3f800000u) - 1.0f;   // [0,1)
        float u = fmaxf(tiny, f * (1.0f - tiny) + tiny);
        float gum = -logf(-logf(u));
        float s = logit + gum;
        if (s > best) { best = s; arg = k; }    // first max on ties, like jnp.argmax
    }
    actions[i] = (float)arg - 1.0f;
}

// ---------------------------------------------------------------------------
extern "C" void kernel_launch(void* const* d_in, const int* in_sizes, int n_in,
                              void* d_out, int out_size, void* d_ws, size_t ws_size,
                              hipStream_t stream) {
    const float* adj = (const float*)d_in[0];
    const float* x   = (const float*)d_in[1];
    const float* h   = (const float*)d_in[2];
    const float* c   = (const float*)d_in[3];
    const float* W1  = (const float*)d_in[4];
    const float* b1  = (const float*)d_in[5];
    const float* uW1 = (const float*)d_in[6];
    const float* ub1 = (const float*)d_in[7];
    const float* Wih = (const float*)d_in[8];
    const float* Whh = (const float*)d_in[9];
    const float* bih = (const float*)d_in[10];
    const float* bhh = (const float*)d_in[11];
    const float* W3  = (const float*)d_in[12];
    const float* b3  = (const float*)d_in[13];

    float* out     = (float*)d_out;
    float* actions = out;                 // [N]
    float* h_out   = out + NN;            // [N,32]
    float* c_out   = out + NN + NN * 32;  // [N,32]

    // ---- carve workspace (~290 MB) ----
    char* p = (char*)d_ws;
    auto carve = [&](size_t bytes) -> char* {
        char* r = p; p += (bytes + 255) & ~(size_t)255; return r;
    };
    bf16*  a16     = (bf16*) carve((size_t)NN * NN * 2);
    bf16*  at16    = (bf16*) carve((size_t)NN * NN * 2);
    float* rowpart = (float*)carve((size_t)64 * NN * 4);
    float* colpart = (float*)carve((size_t)64 * NN * 4);
    float* rowsum  = (float*)carve(NN * 4);
    float* colsum  = (float*)carve(NN * 4);
    float* pA      = (float*)carve((size_t)KSPLIT * NN * 32 * 4);
    float* pB      = (float*)carve((size_t)KSPLIT * NN * 32 * 4);
    float* l1part  = (float*)carve((size_t)(NN / 32) * 96 * 4);
    float* u2part  = (float*)carve((size_t)(NN / 32) * 16 * 4);
    float* u1buf   = (float*)carve(16 * 4);
    float* u2buf   = (float*)carve(16 * 4);
    bf16*  panX    = (bf16*) carve((size_t)NN * 32 * 2);
    bf16*  panV1   = (bf16*) carve((size_t)NN * 32 * 2);
    bf16*  panV2   = (bf16*) carve((size_t)NN * 16 * 2);
    float* Ax      = (float*)carve((size_t)NN * 32 * 4);
    float* ATx     = (float*)carve((size_t)NN * 32 * 4);
    float* v1f     = (float*)carve((size_t)NN * 32 * 4);
    float* Av1     = (float*)carve((size_t)NN * 32 * 4);
    float* ATv1    = (float*)carve((size_t)NN * 32 * 4);
    float* v2f     = (float*)carve((size_t)NN * 16 * 4);
    float* Av2     = (float*)carve((size_t)NN * 16 * 4);
    float* ATv2    = (float*)carve((size_t)NN * 16 * 4);

    const size_t lds_prep  = 128 * 129 * sizeof(float);
    const size_t lds_g2    = (size_t)LKB * 2 * 32 * 16 * sizeof(bf16);  // 64 KB
    const size_t lds_g1    = (size_t)LKB * 1 * 32 * 16 * sizeof(bf16);  // 32 KB
    const size_t lds_lstm  = (128 * 112 + 128 * 32 + 128 + 16) * sizeof(float);

    // 1) single f32 sweep of adj: tile sums + bf16 convert + transpose
    prep_kernel<<<dim3(64, 64), 256, lds_prep, stream>>>(adj, a16, at16, rowpart, colpart);
    sum64_kernel<<<NN / 256, 256, 0, stream>>>(rowpart, rowsum);
    sum64_kernel<<<NN / 256, 256, 0, stream>>>(colpart, colsum);

    // 2) pass 1: [A@x, AT@x]
    pack_panel<<<(NN * 32) / 256, 256, 0, stream>>>(x, panX, 32);
    gemm_kernel<2><<<dim3(64, 2, KSPLIT), 256, lds_g2, stream>>>(a16, at16, panX, pA, pB);
    reduce_scale<<<dim3((NN * 32) / 256, 2), 256, 0, stream>>>(pA, pB, rowsum, colsum, Ax, ATx, 32);

    // 3) layer 1 + global feature u1
    layer1_kernel<<<NN / 256, 256, 0, stream>>>(Ax, ATx, x, W1, b1, v1f, l1part);
    u1_kernel<<<1, 96, 0, stream>>>(l1part, uW1, ub1, u1buf);

    // 4) pass 2: [A@v1, AT@v1]
    pack_panel<<<(NN * 32) / 256, 256, 0, stream>>>(v1f, panV1, 32);
    gemm_kernel<2><<<dim3(64, 2, KSPLIT), 256, lds_g2, stream>>>(a16, at16, panV1, pA, pB);
    reduce_scale<<<dim3((NN * 32) / 256, 2), 256, 0, stream>>>(pA, pB, rowsum, colsum, Av1, ATv1, 32);

    // 5) LSTM (weights in 74 KB LDS)
    lstm_kernel<<<NN / 256, 256, lds_lstm, stream>>>(u1buf, Av1, ATv1, v1f, h, c,
                                                     Wih, Whh, bih, bhh,
                                                     h_out, c_out, v2f, u2part);
    u2_kernel<<<1, 16, 0, stream>>>(u2part, u2buf);

    // 6) pass 3: [A@v2, AT@v2]
    pack_panel<<<(NN * 16) / 256, 256, 0, stream>>>(v2f, panV2, 16);
    gemm_kernel<1><<<dim3(64, 2, KSPLIT), 256, lds_g1, stream>>>(a16, at16, panV2, pA, pB);
    reduce_scale<<<dim3((NN * 16) / 256, 2), 256, 0, stream>>>(pA, pB, rowsum, colsum, Av2, ATv2, 16);

    // 7) layer 3 + tanh/softmax-equivalent Gumbel-argmax sampling
    layer3_kernel<<<NN / 256, 256, 0, stream>>>(u2buf, Av2, ATv2, v2f, W3, b3, actions);
}
